// SelfMutualAttention_41223096107125
// MI455X (gfx1250) — compile-verified, matrix operand order checked by reading, not run on hardware
//
#include <hip/hip_runtime.h>
#include <hip/hip_fp16.h>
#include <math.h>

typedef __attribute__((ext_vector_type(16))) _Float16 v16h;
typedef __attribute__((ext_vector_type(8)))  _Float16 h8;
typedef __attribute__((ext_vector_type(8)))  float    v8f;
typedef __attribute__((ext_vector_type(4)))  float    f4;

#define NTOT   5184   // 72*72 tokens
#define CCH    256    // channels
#define NHEADS 8
#define HD     32     // head dim
#define SCALE  0.17677669529663687f  // 32^-0.5
#define XROW   264    // padded LDS row stride in halves (528 B, 16B-aligned)

__device__ __forceinline__ v8f wmma_f16(v16h a, v16h b, v8f c) {
  return __builtin_amdgcn_wmma_f32_16x16x32_f16(false, a, false, b, (short)0, c,
                                                false, false);
}

// ---- K-contiguous fragment loaders -----------------------------------------
// CDNA5 f16 A/B 16x32 layout: lane L holds row/col (L&15), k set
// {kb..kb+7} U {16+kb..16+kb+7} with kb = (L>=16)?8:0  ->  two 16B chunks.
// v16h element e <-> k = (e<8 ? e : e+8) + kb.
__device__ __forceinline__ v16h load_frag_k1_f16(const _Float16* src, int stride) {
  const int lane = threadIdx.x & 31;
  const int kb   = (lane >> 4) << 3;
  const _Float16* p = src + (size_t)(lane & 15) * stride + kb;
  h8 lo = *(const h8*)(p);
  h8 hi = *(const h8*)(p + 16);
  v16h r;
#pragma unroll
  for (int e = 0; e < 8; ++e) { r[e] = lo[e]; r[e + 8] = hi[e]; }
  return r;
}

__device__ __forceinline__ v16h load_frag_k1_f32(const float* src, int stride) {
  const int lane = threadIdx.x & 31;
  const int kb   = (lane >> 4) << 3;
  const float* p = src + (size_t)(lane & 15) * stride + kb;
  f4 a0 = *(const f4*)(p);
  f4 a1 = *(const f4*)(p + 4);
  f4 b0 = *(const f4*)(p + 16);
  f4 b1 = *(const f4*)(p + 20);
  v16h r;
#pragma unroll
  for (int e = 0; e < 4; ++e) {
    r[e]      = (_Float16)a0[e];
    r[e + 4]  = (_Float16)a1[e];
    r[e + 8]  = (_Float16)b0[e];
    r[e + 12] = (_Float16)b1[e];
  }
  return r;
}

// ---------------------------------------------------------------------------
// Kernel 1: fused QKV projection.  Y[n,o] = sum_c X[c,n]*W[o,c] + b[o]
// x tile transposed into LDS once per block (coalesced), then A-frags are
// ds_load_b128 pairs; weight B-frags are float4 global loads.
// Two independent WMMA accumulator chains (even/odd K-step) hide the f16-WMMA
// RAW hazard gaps.  Q (scaled) / K stored f16 [N][256]; V stored TRANSPOSED
// f16 [256][N] so the attention kernel's V^T operand is K-contiguous.
// grid (N/16, 4, 3), block 128.
// ---------------------------------------------------------------------------
__global__ void qkv_proj_kernel(const float* __restrict__ x1, const float* __restrict__ x2,
                                const float* __restrict__ Wq, const float* __restrict__ bq,
                                const float* __restrict__ Wk, const float* __restrict__ bk,
                                const float* __restrict__ Wv, const float* __restrict__ bv,
                                _Float16* __restrict__ qb, _Float16* __restrict__ kbuf,
                                _Float16* __restrict__ vbT) {
  __shared__ _Float16 xt[16 * XROW];
  const int tid  = threadIdx.x;
  const int wave = tid >> 5;
  const int lane = tid & 31;
  const int n0 = blockIdx.x * 16;
  const int o0 = blockIdx.y * 64 + wave * 16;
  const int which = blockIdx.z;

  const float* X    = (which == 0) ? x1 : x2;
  const float* W    = (which == 0) ? Wq : (which == 1) ? Wk : Wv;
  const float* bias = (which == 0) ? bq : (which == 1) ? bk : bv;

  // stage x tile transposed: xt[n_local][c] f16, coalesced reads of x[c, n0+j]
  {
    const int j = tid & 15;
    for (int c = tid >> 4; c < CCH; c += 8)
      xt[j * XROW + c] = (_Float16)X[(size_t)c * NTOT + n0 + j];
  }
  __syncthreads();

  v8f acc0 = {}, acc1 = {};            // two independent WMMA chains
#pragma unroll
  for (int k0 = 0; k0 < CCH; k0 += 64) {
    v16h a0 = load_frag_k1_f16(xt + k0, XROW);
    v16h b0 = load_frag_k1_f32(W + (size_t)o0 * CCH + k0, CCH);
    v16h a1 = load_frag_k1_f16(xt + k0 + 32, XROW);
    v16h b1 = load_frag_k1_f32(W + (size_t)o0 * CCH + k0 + 32, CCH);
    acc0 = wmma_f16(a0, b0, acc0);
    acc1 = wmma_f16(a1, b1, acc1);
  }

  const int   q  = lane & 15;
  const int   mh = (lane >> 4) << 3;
  const float bb = bias[o0 + q];
  if (which == 2) {            // V transposed: 8 contiguous halves -> one 16B store
    h8 pk;
#pragma unroll
    for (int r = 0; r < 8; ++r) pk[r] = (_Float16)(acc0[r] + acc1[r] + bb);
    *(h8*)(vbT + (size_t)(o0 + q) * NTOT + n0 + mh) = pk;
  } else {
    _Float16* out = (which == 0) ? qb : kbuf;
    const float scl = (which == 0) ? SCALE : 1.0f;
#pragma unroll
    for (int r = 0; r < 8; ++r)
      out[(size_t)(n0 + r + mh) * CCH + o0 + q] =
          (_Float16)((acc0[r] + acc1[r] + bb) * scl);
  }
}

// ---------------------------------------------------------------------------
// Kernel 2: flash attention. One wave owns 32 queries of one head.
// S^T = K*Q^T so query = lane column; per-query softmax stats need only an
// in-lane reduction + one shfl_xor(16).  The S^T D-layout coincides with the
// P^T B-operand layout, so bP is built purely in registers (no LDS).
// O^T = V^T * P^T with V^T read K-contiguously from the transposed V buffer.
// grid (N/64, NHEADS), block 64 (2 waves).  8 WMMAs per 32-key tile across
// 4 independent O-accumulator chains.
// ---------------------------------------------------------------------------
__device__ __forceinline__ v16h softmax_update(v8f s0, v8f s1, float& mrun,
                                               float& lrun, float& alpha_out) {
  float mloc = -INFINITY;
#pragma unroll
  for (int r = 0; r < 8; ++r) {
    mloc = fmaxf(mloc, s0[r]);
    mloc = fmaxf(mloc, s1[r]);
  }
  mloc = fmaxf(mloc, __shfl_xor(mloc, 16, 32));     // merge key halves
  const float mnew  = fmaxf(mrun, mloc);
  const float alpha = __expf(mrun - mnew);
  v16h bP;
  float lloc = 0.0f;
#pragma unroll
  for (int r = 0; r < 8; ++r) {
    const float p0 = __expf(s0[r] - mnew);          // key = r + half
    const float p1 = __expf(s1[r] - mnew);          // key = 16 + r + half
    lloc += p0 + p1;
    bP[r]     = (_Float16)p0;                       // B elem e<8  -> k = e+kb
    bP[r + 8] = (_Float16)p1;                       // B elem e>=8 -> k = 16+(e-8)+kb
  }
  lloc += __shfl_xor(lloc, 16, 32);
  lrun = lrun * alpha + lloc;
  mrun = mnew;
  alpha_out = alpha;
  return bP;
}

__global__ void flash_attn_kernel(const _Float16* __restrict__ qb,
                                  const _Float16* __restrict__ kbuf,
                                  const _Float16* __restrict__ vbT,
                                  float* __restrict__ ob) {
  const int wave = threadIdx.x >> 5;
  const int lane = threadIdx.x & 31;
  const int h  = blockIdx.y;
  const int n0 = blockIdx.x * 64 + wave * 32;       // 32 queries per wave

  const _Float16* qp = qb + (size_t)n0 * CCH + h * HD;
  const v16h bQ0 = load_frag_k1_f16(qp, CCH);
  const v16h bQ1 = load_frag_k1_f16(qp + 16 * CCH, CCH);
  const _Float16* vrow = vbT + (size_t)h * HD * NTOT;

  v8f o00 = {}, o01 = {}, o10 = {}, o11 = {};       // [qtile][d-half]
  float m0 = -INFINITY, m1 = -INFINITY, l0 = 0.0f, l1 = 0.0f;

  for (int kt = 0; kt < NTOT; kt += 32) {
    const _Float16* kp = kbuf + (size_t)kt * CCH + h * HD;
    if (kt + 32 < NTOT) {
      __builtin_prefetch(kp + 32 * CCH, 0, 1);
      __builtin_prefetch(vrow + kt + 32, 0, 1);
    }

    v16h aK0 = load_frag_k1_f16(kp, CCH);           // keys kt..kt+15
    v16h aK1 = load_frag_k1_f16(kp + 16 * CCH, CCH);
    v8f z = {};
    v8f s00 = wmma_f16(aK0, bQ0, z);
    v8f s10 = wmma_f16(aK1, bQ0, z);
    v8f s01 = wmma_f16(aK0, bQ1, z);
    v8f s11 = wmma_f16(aK1, bQ1, z);

    v16h aV0 = load_frag_k1_f16(vrow + kt, NTOT);              // d 0-15
    v16h aV1 = load_frag_k1_f16(vrow + (size_t)16 * NTOT + kt, NTOT);  // d 16-31

    float a0, a1;
    v16h bP0 = softmax_update(s00, s10, m0, l0, a0);
#pragma unroll
    for (int r = 0; r < 8; ++r) { o00[r] *= a0; o01[r] *= a0; }
    o00 = wmma_f16(aV0, bP0, o00);
    o01 = wmma_f16(aV1, bP0, o01);

    v16h bP1 = softmax_update(s01, s11, m1, l1, a1);
#pragma unroll
    for (int r = 0; r < 8; ++r) { o10[r] *= a1; o11[r] *= a1; }
    o10 = wmma_f16(aV0, bP1, o10);
    o11 = wmma_f16(aV1, bP1, o11);
  }

  const int q    = lane & 15;
  const int half = (lane >> 4) << 3;
  const float i0 = 1.0f / l0, i1 = 1.0f / l1;
  // O^T elem (d = half+r [+16], q): 8 consecutive f32 per frag -> b128 stores
  {
    float* op = ob + (size_t)(n0 + q) * CCH + h * HD + half;
    f4 w0, w1, w2, w3;
#pragma unroll
    for (int r = 0; r < 4; ++r) {
      w0[r] = o00[r] * i0; w1[r] = o00[r + 4] * i0;
      w2[r] = o01[r] * i0; w3[r] = o01[r + 4] * i0;
    }
    *(f4*)(op)      = w0; *(f4*)(op + 4)  = w1;
    *(f4*)(op + 16) = w2; *(f4*)(op + 20) = w3;
  }
  {
    float* op = ob + (size_t)(n0 + 16 + q) * CCH + h * HD + half;
    f4 w0, w1, w2, w3;
#pragma unroll
    for (int r = 0; r < 4; ++r) {
      w0[r] = o10[r] * i1; w1[r] = o10[r + 4] * i1;
      w2[r] = o11[r] * i1; w3[r] = o11[r + 4] * i1;
    }
    *(f4*)(op)      = w0; *(f4*)(op + 4)  = w1;
    *(f4*)(op + 16) = w2; *(f4*)(op + 20) = w3;
  }
}

// ---------------------------------------------------------------------------
// Kernel 3: output projection. out[o,n] = sum_c O[n,c]*Wo[o,c] + bo[o]
// Both operands K-contiguous -> all-vector loads; two independent WMMA chains;
// contiguous b128 stores.  grid (N/16, 4), block 128.
// ---------------------------------------------------------------------------
__global__ void out_proj_kernel(const float* __restrict__ ob,
                                const float* __restrict__ Wo,
                                const float* __restrict__ bo,
                                float* __restrict__ out) {
  const int wave = threadIdx.x >> 5;
  const int lane = threadIdx.x & 31;
  const int n0 = blockIdx.x * 16;
  const int o0 = blockIdx.y * 64 + wave * 16;

  v8f acc0 = {}, acc1 = {};
#pragma unroll
  for (int k0 = 0; k0 < CCH; k0 += 64) {
    v16h a0 = load_frag_k1_f32(ob + (size_t)n0 * CCH + k0, CCH);
    v16h b0 = load_frag_k1_f32(Wo + (size_t)o0 * CCH + k0, CCH);
    v16h a1 = load_frag_k1_f32(ob + (size_t)n0 * CCH + k0 + 32, CCH);
    v16h b1 = load_frag_k1_f32(Wo + (size_t)o0 * CCH + k0 + 32, CCH);
    acc0 = wmma_f16(a0, b0, acc0);
    acc1 = wmma_f16(a1, b1, acc1);
  }
  const int   q  = lane & 15;
  const int   mh = (lane >> 4) << 3;
  const float bb = bo[o0 + q];
  f4 w0, w1;
#pragma unroll
  for (int r = 0; r < 4; ++r) {
    w0[r] = acc0[r] + acc1[r] + bb;
    w1[r] = acc0[r + 4] + acc1[r + 4] + bb;
  }
  float* op = out + (size_t)(o0 + q) * NTOT + n0 + mh;
  *(f4*)(op)     = w0;
  *(f4*)(op + 4) = w1;
}

// ---------------------------------------------------------------------------
extern "C" void kernel_launch(void* const* d_in, const int* in_sizes, int n_in,
                              void* d_out, int out_size, void* d_ws, size_t ws_size,
                              hipStream_t stream) {
  const float* x1 = (const float*)d_in[0];
  const float* x2 = (const float*)d_in[1];
  const float* Wq = (const float*)d_in[2];
  const float* bq = (const float*)d_in[3];
  const float* Wk = (const float*)d_in[4];
  const float* bk = (const float*)d_in[5];
  const float* Wv = (const float*)d_in[6];
  const float* bv = (const float*)d_in[7];
  const float* Wo = (const float*)d_in[8];
  const float* bo = (const float*)d_in[9];
  float* out = (float*)d_out;

  // workspace: Q/K f16 [N][256], V^T f16 [256][N], O f32 [N][256]  (~13.3 MB)
  char* ws = (char*)d_ws;
  const size_t fh = (size_t)NTOT * CCH * sizeof(_Float16);
  _Float16* qb  = (_Float16*)(ws);
  _Float16* kb  = (_Float16*)(ws + fh);
  _Float16* vbT = (_Float16*)(ws + 2 * fh);
  float*    obf = (float*)(ws + 3 * fh);

  qkv_proj_kernel<<<dim3(NTOT / 16, 4, 3), 128, 0, stream>>>(
      x1, x2, Wq, bq, Wk, bk, Wv, bv, qb, kb, vbT);
  flash_attn_kernel<<<dim3(NTOT / 64, NHEADS), 64, 0, stream>>>(qb, kb, vbT, obf);
  out_proj_kernel<<<dim3(NTOT / 16, 4), 128, 0, stream>>>(obf, Wo, bo, out);
}
